// Node2Vec_42047729828085
// MI455X (gfx1250) — matrix operand order repaired and checked by reading, not compiled, not run on hardware
//
#include <hip/hip_runtime.h>
#include <hip/hip_bf16.h>
#include <stdint.h>

#define N_NODES      10000
#define DEGREE       32
#define BATCH        128
#define EMBED_DIM    128
#define WALK_LENGTH  20
#define CONTEXT_SIZE 10
#define NUM_W        (1 + WALK_LENGTH + 1 - CONTEXT_SIZE)  /* 12 */
#define WALK_NODES   (WALK_LENGTH + 1)                     /* 21 */
#define P_FACTOR     0.5f   /* clip(1-P,0,1) */
#define Q_CLAMP      0.8f   /* clip(Q,0,1)  */
#define LOSS_EPS     1e-15f

typedef __bf16 bf16_t;
typedef __attribute__((ext_vector_type(16))) __bf16 v16bf;
typedef __attribute__((ext_vector_type(8)))  __bf16 v8bf;
typedef __attribute__((ext_vector_type(8)))  float  v8f;

// ---------------------------------------------------------------- RNG helpers
__device__ __forceinline__ uint32_t hash_u32(uint32_t x) {
    x ^= x >> 16; x *= 0x7feb352dU;
    x ^= x >> 15; x *= 0x846ca68bU;
    x ^= x >> 16;
    return x;
}
__device__ __forceinline__ uint32_t hash3(uint32_t a, uint32_t b, uint32_t c) {
    return hash_u32(a * 0x9E3779B1u ^ b * 0x85EBCA77u ^ c * 0xC2B2AE3Du ^ 0x42424242u);
}
__device__ __forceinline__ float u01(uint32_t h) {
    return (float)(h >> 8) * (1.0f / 16777216.0f);   // [0,1)
}

// fast sigmoid: v_exp_f32 + v_add + v_rcp_f32 (no IEEE divide expansion)
__device__ __forceinline__ float fast_sigmoid(float x) {
    return __builtin_amdgcn_rcpf(1.0f + __expf(-x));
}

// ------------------------------------------------------- f32 -> bf16 convert
__global__ void n2v_cvt_emb(const float* __restrict__ src,
                            bf16_t* __restrict__ dst, int n) {
    int i = blockIdx.x * blockDim.x + threadIdx.x;
    if (i < n) dst[i] = (bf16_t)src[i];
}

// ------------------------------------------------- wave-per-walker sampling
// lane l = candidate edge l of the current node (DEGREE == warpSize == 32)
__global__ void n2v_walk(const long long* __restrict__ rowptr,
                         const long long* __restrict__ col,
                         const long long* __restrict__ row,
                         const float*     __restrict__ weight,
                         const long long* __restrict__ batch,
                         int* __restrict__ pos_walks,
                         int* __restrict__ neg_walks) {
    const int lane = threadIdx.x & 31;
    const int wave = threadIdx.x >> 5;
    const int b = blockIdx.x * (blockDim.x >> 5) + wave;
    if (b >= BATCH) return;

    const long long start_node = batch[b];
    int cur = (int)start_node;
    long long prev_loc = start_node;   // reference quirk: node id used as edge index

    if (lane == 0) {
        pos_walks[b * WALK_NODES] = cur;
        neg_walks[b * WALK_NODES] = cur;
    }
    if (lane < WALK_LENGTH) {
        uint32_t h = hash3(0xD00Du, (uint32_t)b, (uint32_t)lane);
        neg_walks[b * WALK_NODES + 1 + lane] = (int)(h % N_NODES);
    }

    for (int step = 0; step < WALK_LENGTH; ++step) {
        const long long e0 = rowptr[cur];
        const long long e  = e0 + lane;
        float w = weight[e];
        const long long c = col[e];
        const long long r = row[e];
        const long long prev_row = row[prev_loc];
        const long long prev_col = col[prev_loc];
        if (r == prev_col && c == prev_row) w *= P_FACTOR;

        // inclusive wave32 prefix sum -> CDF
        float s = w;
        #pragma unroll
        for (int off = 1; off < 32; off <<= 1) {
            float t = __shfl_up(s, off);
            if (lane >= off) s += t;
        }
        const float total = __shfl(s, 31);
        const float rnd = u01(hash3((uint32_t)step, (uint32_t)b, 0x5E1Fu));
        unsigned long long m = __ballot(rnd * total < s);
        const int sel = (m == 0ull) ? 31 : (__ffsll(m) - 1);
        const long long loc = e0 + sel;
        const int pos = (int)col[loc];

        if (lane == 0) pos_walks[b * WALK_NODES + 1 + step] = pos;

        const float q_rnd = u01(hash3((uint32_t)step, (uint32_t)b, 0x0FADu));
        cur = (q_rnd < Q_CLAMP) ? pos : cur;
        prev_loc = loc;
    }
}

// ----------------------------------------------------------- zero the output
__global__ void n2v_zero(float* __restrict__ out, int n) {
    int i = blockIdx.x * blockDim.x + threadIdx.x;
    if (i < n) out[i] = 0.0f;
}

// ---------------------------------------------- Gram-matrix loss via WMMA
// One wave per (walk, sign). G = E_walk (21x128, zero-padded to 32x128) x E^T.
// Two 16x16 f32 accumulator tiles: rows 0..15, cols 0..15 and 16..31.
// K = 128 split into 4 chunks of 32 -> 8 x v_wmma_f32_16x16x32_bf16 per wave.
__global__ void n2v_gram_loss(const bf16_t* __restrict__ emb,
                              const int* __restrict__ pos_walks,
                              const int* __restrict__ neg_walks,
                              float* __restrict__ out) {
    const int lane = threadIdx.x & 31;
    const int wave = threadIdx.x >> 5;
    const int gid  = blockIdx.x * (blockDim.x >> 5) + wave;   // 0 .. 2*BATCH-1
    if (gid >= 2 * BATCH) return;
    const bool positive = (gid < BATCH);
    const int b = positive ? gid : gid - BATCH;
    const int* __restrict__ walk =
        (positive ? pos_walks : neg_walks) + b * WALK_NODES;

    const int m    = lane & 15;   // A row / B column within tile
    const int half = lane >> 4;   // K-half select per WMMA VGPR layout

    const int nodeA  = walk[m];                                  // M = m row
    const int nodeB1 = (16 + m < WALK_NODES) ? walk[16 + m] : 0; // N = 16+m col

    const bf16_t* __restrict__ rowA  = emb + (long)nodeA  * EMBED_DIM;
    const bf16_t* __restrict__ rowB1 = emb + (long)nodeB1 * EMBED_DIM;
    __builtin_prefetch(rowA,  0, 1);
    __builtin_prefetch(rowB1, 0, 1);

    v8f c0 = {};
    v8f c1 = {};
    #pragma unroll
    for (int kc = 0; kc < 4; ++kc) {
        const int kbase = kc * 32;
        // A (16x32 bf16): lane holds row m; half 0 -> K 0..15,
        // half 1 -> K 16..31 of this chunk.
        v16bf a = *(const v16bf*)(rowA + kbase + half * 16);

        // B (32x16 bf16): lane holds column (lane&15); half selects K 0..15 /
        // 16..31 of the chunk (16 contiguous bf16).
        v16bf b0 = *(const v16bf*)(rowA  + kbase + half * 16);  // tile0 col = walk[m]
        v16bf b1 = *(const v16bf*)(rowB1 + kbase + half * 16);  // tile1 col = walk[16+m]

        c0 = __builtin_amdgcn_wmma_f32_16x16x32_bf16(
                 false, a, false, b0, (short)0, c0, false, false);
        c1 = __builtin_amdgcn_wmma_f32_16x16x32_bf16(
                 false, a, false, b1, (short)0, c1, false, false);
    }

    // Valid loss entries: G[j, j+1 .. j+CONTEXT_SIZE-1], j in 0..NUM_W-1.
    // C/D layout: lane holds (M = half*8 + r, N = tile*16 + (lane&15)).
    const float inv = 1.0f / (float)(NUM_W * BATCH * (CONTEXT_SIZE - 1));
    float acc = 0.0f;
    #pragma unroll
    for (int r = 0; r < 8; ++r) {
        const int M = half * 8 + r;
        {
            const int N = m;
            if (M < NUM_W && N > M && N <= M + (CONTEXT_SIZE - 1)) {
                const float sig = fast_sigmoid(c0[r]);
                const float tgt = positive ? sig : (1.0f - sig);
                acc -= __logf(tgt + LOSS_EPS);
            }
        }
        {
            const int N = 16 + m;
            if (M < NUM_W && N > M && N <= M + (CONTEXT_SIZE - 1)) {
                const float sig = fast_sigmoid(c1[r]);
                const float tgt = positive ? sig : (1.0f - sig);
                acc -= __logf(tgt + LOSS_EPS);
            }
        }
    }

    // wave32 reduction + one atomic per wave
    #pragma unroll
    for (int off = 16; off > 0; off >>= 1) acc += __shfl_down(acc, off);
    if (lane == 0) atomicAdd(out, acc * inv);
}

// --------------------------------------------------------------------- launch
extern "C" void kernel_launch(void* const* d_in, const int* in_sizes, int n_in,
                              void* d_out, int out_size, void* d_ws, size_t ws_size,
                              hipStream_t stream) {
    const long long* rowptr = (const long long*)d_in[0];
    const long long* col    = (const long long*)d_in[1];
    const long long* row    = (const long long*)d_in[2];
    const float*     weight = (const float*)d_in[3];
    const long long* batch  = (const long long*)d_in[4];
    const float*     emb_f  = (const float*)d_in[5];
    float* out = (float*)d_out;

    // workspace layout
    char* ws = (char*)d_ws;
    bf16_t* emb_bf = (bf16_t*)ws;
    size_t off = (size_t)N_NODES * EMBED_DIM * sizeof(bf16_t);
    off = (off + 255) & ~(size_t)255;
    int* pos_walks = (int*)(ws + off);
    off += (size_t)BATCH * WALK_NODES * sizeof(int);
    off = (off + 255) & ~(size_t)255;
    int* neg_walks = (int*)(ws + off);

    const int nEmb = N_NODES * EMBED_DIM;
    hipLaunchKernelGGL(n2v_cvt_emb, dim3((nEmb + 255) / 256), dim3(256), 0, stream,
                       emb_f, emb_bf, nEmb);
    hipLaunchKernelGGL(n2v_walk, dim3(BATCH / 8), dim3(256), 0, stream,
                       rowptr, col, row, weight, batch, pos_walks, neg_walks);
    hipLaunchKernelGGL(n2v_zero, dim3(1), dim3(32), 0, stream, out, out_size);
    hipLaunchKernelGGL(n2v_gram_loss, dim3((2 * BATCH) / 8), dim3(256), 0, stream,
                       emb_bf, pos_walks, neg_walks, out);
}
